// JetAssignmentTransformer_82128364634206
// MI455X (gfx1250) — matrix operand order repaired and checked by reading, not compiled, not run on hardware
//
#include <hip/hip_runtime.h>
#include <hip/hip_bf16.h>
#include <math.h>

// ---------------- model constants ----------------
#define BATCH_   8192
#define SJ       6      // jets per event
#define DM       128    // d_model
#define NHEAD_   8
#define HD_      16     // head dim (main)
#define DFF_     256
#define NA_      10     // NUM_ASSIGN
#define NG_      20     // 2 * NUM_ASSIGN
#define GS_      4      // group seq len
#define GNH_     4
#define GHD_     32
#define FEATD    410
#define FEATP    416    // padded feature width (K multiple of 32)
#define KPAD_PHYS 32    // padded K for phys_proj (10 -> 32)
#define NCHUNK   8
#define CB       (BATCH_ / NCHUNK)

typedef __attribute__((ext_vector_type(16))) _Float16 v16h;
typedef __attribute__((ext_vector_type(8)))  _Float16 v8h;
typedef __attribute__((ext_vector_type(4)))  _Float16 v4h;
typedef __attribute__((ext_vector_type(8)))  float    v8f;

// rows 0..9 = G1 assignments (0,i,j), rows 10..19 = G2 complements
__device__ const int JET_TABLE[NG_][3] = {
  {0,1,2},{0,1,3},{0,1,4},{0,1,5},{0,2,3},{0,2,4},{0,2,5},{0,3,4},{0,3,5},{0,4,5},
  {3,4,5},{2,4,5},{2,3,5},{2,3,4},{1,4,5},{1,3,5},{1,3,4},{1,2,5},{1,2,4},{1,2,3}};

__device__ __forceinline__ float gelu_f(float x) {
  return 0.5f * x * (1.0f + erff(x * 0.7071067811865475f));
}
__device__ __forceinline__ float wrap_dphi(float d) {
  return d - 6.2831853071795862f * rintf(d * 0.15915494309189535f);
}

// ---------------------------------------------------------------------------
// f16 fragment loaders (CDNA5 ISA 7.12.2 layouts), pure 16-byte loads.
// A fragment (16x32, lane m = lane&15, K-half = lane>>4):
//   elems 0..7  -> K = kb + half*8 + e
//   elems 8..15 -> K = kb + 16 + half*8 + (e-8)
// B fragment (32x16, lane n = lane&15, K-half = lane>>4):
//   elems 0..7  -> K = kb + half*8 + e
//   elems 8..15 -> K = kb + 16 + half*8 + (e-8)
// ---------------------------------------------------------------------------
__device__ __forceinline__ v16h load_afrag(const _Float16* __restrict__ Ap,
                                           int kb, int half) {
  v8h a0 = *(const v8h*)(Ap + kb + half * 8);
  v8h a1 = *(const v8h*)(Ap + kb + 16 + half * 8);
  v16h r;
  #pragma unroll
  for (int e = 0; e < 8; ++e) { r[e] = a0[e]; r[8 + e] = a1[e]; }
  return r;
}
__device__ __forceinline__ v16h load_bfrag(const _Float16* __restrict__ Wp,
                                           int kb, int half) {
  v8h b0 = *(const v8h*)(Wp + kb + half * 8);
  v8h b1 = *(const v8h*)(Wp + kb + 16 + half * 8);
  v16h r;
  #pragma unroll
  for (int e = 0; e < 8; ++e) { r[e] = b0[e]; r[8 + e] = b1[e]; }
  return r;
}

// ---------------------------------------------------------------------------
// Aligned WMMA GEMM: C[M,N] = act(A[M,K]h @ W[N,K]h^T + bias [+ R])
// Requirements (guaranteed by caller): M%16==0, N%128==0, K%32==0,
// lda%8==0, ldw%8==0.  One wave computes a 16x128 strip: one A fragment
// feeds 8 v_wmma_f32_16x16x32_f16.  Software-pipelined: the next B (and at
// strip boundaries the next A) fragment is issued before each WMMA so loads
// stay in flight under the matrix op instead of s_wait_loadcnt 0.
// ---------------------------------------------------------------------------
template<int ACT, bool RESID, typename OutT>
__global__ __launch_bounds__(256)
void k_gemm(const _Float16* __restrict__ A, int lda,
            const _Float16* __restrict__ W, int ldw,
            const float* __restrict__ bias,
            const float* __restrict__ R,
            OutT* __restrict__ C, int ldc,
            long M, int N, int K)
{
  const int lane = threadIdx.x & 31;
  const int wv   = threadIdx.x >> 5;
  const int groupsN = N >> 7;                 // N / 128
  const long tasks  = (M >> 4) * (long)groupsN;
  const long task   = (long)blockIdx.x * 8 + wv;
  if (task >= tasks) return;
  const long tm = task / groupsN;
  const int  tg = (int)(task % groupsN);
  const int half = lane >> 4;
  const int l15  = lane & 15;

  const _Float16* __restrict__ Ap = A + (tm * 16 + l15) * (long)lda;
  const _Float16* __restrict__ Wp = W + ((long)tg * 128 + l15) * (long)ldw;

  v8f acc[8] = {};
  v16h a = load_afrag(Ap, 0, half);
  v16h b = load_bfrag(Wp, 0, half);
  for (int kb = 0; kb < K; kb += 32) {
    #pragma unroll
    for (int t = 0; t < 8; ++t) {
      v16h a_cur = a, b_cur = b;
      if (t < 7) {                         // prefetch next B of this strip
        b = load_bfrag(Wp + (long)(t + 1) * 16 * ldw, kb, half);
      } else if (kb + 32 < K) {            // prefetch next K-step's A and B
        a = load_afrag(Ap, kb + 32, half);
        b = load_bfrag(Wp, kb + 32, half);
      }
      acc[t] = __builtin_amdgcn_wmma_f32_16x16x32_f16(
          false, a_cur, false, b_cur, (short)0, acc[t], false, false);
    }
  }

  // C element r of tile t -> (row = tm*16 + r + 8*half, col = tg*128+t*16+l15)
  const long rowBase = tm * 16 + 8 * half;
  #pragma unroll
  for (int t = 0; t < 8; ++t) {
    const long col = (long)tg * 128 + t * 16 + l15;
    const float bc = bias ? bias[col] : 0.f;
    #pragma unroll
    for (int r = 0; r < 8; ++r) {
      long row = rowBase + r;
      float v = acc[t][r] + bc;
      if (RESID) v += R[row * (long)ldc + col];
      if (ACT == 1) v = fmaxf(v, 0.f);
      else if (ACT == 2) v = gelu_f(v);
      C[row * (long)ldc + col] = (OutT)v;
    }
  }
}

// ---------------------------------------------------------------------------
// Weight convert + K-pad: dst[r, 0:kd] = f16(src[r, 0:ks]) zero-padded
// ---------------------------------------------------------------------------
__global__ __launch_bounds__(256)
void k_cvt_pad(const float* __restrict__ src, _Float16* __restrict__ dst,
               int rows, int ks, int kd)
{
  long i = (long)blockIdx.x * blockDim.x + threadIdx.x;
  long total = (long)rows * kd;
  if (i >= total) return;
  int c = (int)(i % kd); long r = i / kd;
  dst[i] = (c < ks) ? (_Float16)src[r * (long)ks + c] : (_Float16)0.f;
}

// ---------------------------------------------------------------------------
// Small per-head attention: one thread per (seq, head, q-row), f16 in/out
// ---------------------------------------------------------------------------
template<int S, int NHE, int HDIM, bool HASBIAS>
__global__ __launch_bounds__(256)
void k_attn(const _Float16* __restrict__ qkv,
            const float* __restrict__ logpt,
            const float* __restrict__ biasw,
            _Float16* __restrict__ out, long nSeq)
{
  long t = (long)blockIdx.x * blockDim.x + threadIdx.x;
  long total = nSeq * NHE * S;
  if (t >= total) return;
  int q = (int)(t % S);  long r = t / S;
  int h = (int)(r % NHE); long seq = r / NHE;
  const int D = NHE * HDIM;
  const long base = seq * (long)S * 3 * D;
  const float scale = rsqrtf((float)HDIM);

  float qv[HDIM];
  const _Float16* qp = qkv + base + (long)q * 3 * D + h * HDIM;
  #pragma unroll
  for (int c = 0; c < HDIM / 8; ++c) {
    v8h vv = ((const v8h*)qp)[c];
    #pragma unroll
    for (int e = 0; e < 8; ++e) qv[c * 8 + e] = (float)vv[e];
  }

  float sc[S]; float mx = -1e30f;
  #pragma unroll
  for (int j = 0; j < S; ++j) {
    const _Float16* kp = qkv + base + (long)j * 3 * D + D + h * HDIM;
    float s = 0.f;
    #pragma unroll
    for (int c = 0; c < HDIM / 8; ++c) {
      v8h vv = ((const v8h*)kp)[c];
      #pragma unroll
      for (int e = 0; e < 8; ++e) s += qv[c * 8 + e] * (float)vv[e];
    }
    s *= scale;
    if (HASBIAS) s += (logpt[seq * S + q] - logpt[seq * S + j]) * biasw[h];
    sc[j] = s; mx = fmaxf(mx, s);
  }
  float den = 0.f;
  #pragma unroll
  for (int j = 0; j < S; ++j) { sc[j] = expf(sc[j] - mx); den += sc[j]; }
  float inv = 1.f / den;

  float ov[HDIM];
  #pragma unroll
  for (int d = 0; d < HDIM; ++d) ov[d] = 0.f;
  #pragma unroll
  for (int j = 0; j < S; ++j) {
    const _Float16* vp = qkv + base + (long)j * 3 * D + 2 * D + h * HDIM;
    float pj = sc[j] * inv;
    #pragma unroll
    for (int c = 0; c < HDIM / 8; ++c) {
      v8h vv = ((const v8h*)vp)[c];
      #pragma unroll
      for (int e = 0; e < 8; ++e) ov[c * 8 + e] += pj * (float)vv[e];
    }
  }
  _Float16* op = out + (seq * (long)S + q) * D + h * HDIM;
  #pragma unroll
  for (int c = 0; c < HDIM / 8; ++c) {
    v8h o;
    #pragma unroll
    for (int e = 0; e < 8; ++e) o[e] = (_Float16)ov[c * 8 + e];
    ((v8h*)op)[c] = o;
  }
}

// ---------------------------------------------------------------------------
// LayerNorm over rows of 128 (f32 in, f16 out), one wave per row
// ---------------------------------------------------------------------------
__global__ __launch_bounds__(256)
void k_ln128(const float* __restrict__ X, const float* __restrict__ g,
             const float* __restrict__ b, _Float16* __restrict__ Y, long rows)
{
  long row = (long)blockIdx.x * 8 + (threadIdx.x >> 5);
  int lane = threadIdx.x & 31;
  if (row >= rows) return;
  float4 v = ((const float4*)(X + row * 128))[lane];
  float s = v.x + v.y + v.z + v.w;
  #pragma unroll
  for (int m = 16; m > 0; m >>= 1) s += __shfl_xor(s, m, 32);
  float mu = s * (1.f / 128.f);
  float4 d = {v.x - mu, v.y - mu, v.z - mu, v.w - mu};
  float qq = d.x*d.x + d.y*d.y + d.z*d.z + d.w*d.w;
  #pragma unroll
  for (int m = 16; m > 0; m >>= 1) qq += __shfl_xor(qq, m, 32);
  float rstd = rsqrtf(qq * (1.f / 128.f) + 1e-5f);
  float4 gg = ((const float4*)g)[lane];
  float4 bb = ((const float4*)b)[lane];
  v4h o = {(_Float16)(gg.x * d.x * rstd + bb.x),
           (_Float16)(gg.y * d.y * rstd + bb.y),
           (_Float16)(gg.z * d.z * rstd + bb.z),
           (_Float16)(gg.w * d.w * rstd + bb.w)};
  ((v4h*)(Y + row * 128))[lane] = o;
}

// ---------------------------------------------------------------------------
// Input embedding + log_pt
// ---------------------------------------------------------------------------
__global__ __launch_bounds__(256)
void k_embed(const float* __restrict__ fm, const float* __restrict__ Win,
             const float* __restrict__ bin, const float* __restrict__ pos,
             float* __restrict__ x, float* __restrict__ logpt)
{
  long i = (long)blockIdx.x * blockDim.x + threadIdx.x;
  if (i >= (long)BATCH_ * SJ * DM) return;
  int d = (int)(i % DM); long t = i / DM; int j = (int)(t % SJ);
  const float* p = fm + t * 4;
  const float* w = Win + (long)d * 4;
  x[i] = w[0]*p[0] + w[1]*p[1] + w[2]*p[2] + w[3]*p[3] + bin[d] + pos[j * DM + d];
  if (d == 0) {
    float pt = sqrtf(p[1]*p[1] + p[2]*p[2]);
    logpt[t] = logf(fmaxf(pt, 1e-8f));
  }
}

// ---------------------------------------------------------------------------
// Physics: one thread per (batch, assignment); writes f16 intra rows
// (stride 32, zero-padded) and f32 phys26 with fused phys_norm LayerNorm.
// ---------------------------------------------------------------------------
__global__ __launch_bounds__(128)
void k_physics(const float* __restrict__ fm,
               const float* __restrict__ png, const float* __restrict__ pnb,
               _Float16* __restrict__ intra, float* __restrict__ phys26)
{
  long t = (long)blockIdx.x * blockDim.x + threadIdx.x;
  if (t >= (long)BATCH_ * NA_) return;
  int a = (int)(t % NA_); long b = t / NA_;
  const float* base = fm + b * (long)SJ * 4;
  float E[SJ], px[SJ], py[SJ], pz[SJ];
  #pragma unroll
  for (int j = 0; j < SJ; ++j) {
    E[j] = base[j*4]; px[j] = base[j*4+1]; py[j] = base[j*4+2]; pz[j] = base[j*4+3];
  }
  float feats[26];
  float P[2][4];
  const int PRS[3][2] = {{0,1},{0,2},{1,2}};
  for (int g = 0; g < 2; ++g) {
    const int* J = JET_TABLE[g * NA_ + a];
    float pt[3], eta[3], phi[3];
    float sE = 0, sx = 0, sy = 0, sz = 0;
    #pragma unroll
    for (int i = 0; i < 3; ++i) {
      int j = J[i];
      float ptv = fmaxf(sqrtf(px[j]*px[j] + py[j]*py[j]), 1e-8f);
      pt[i] = ptv; eta[i] = asinhf(pz[j] / ptv); phi[i] = atan2f(py[j], px[j]);
      sE += E[j]; sx += px[j]; sy += py[j]; sz += pz[j];
    }
    P[g][0]=sE; P[g][1]=sx; P[g][2]=sy; P[g][3]=sz;
    float sum = fmaxf(pt[0]+pt[1]+pt[2], 1e-8f);
    float f0 = pt[0]/sum, f1 = pt[1]/sum, f2 = pt[2]/sum, tf;
    if (f0 < f1) { tf=f0; f0=f1; f1=tf; }
    if (f1 < f2) { tf=f1; f1=f2; f2=tf; }
    if (f0 < f1) { tf=f0; f0=f1; f1=tf; }
    float m2g = fmaxf(sE*sE - (sx*sx + sy*sy + sz*sz), 1e-8f);
    float m = sqrtf(m2g);
    float o10[10];
    o10[0]=f0; o10[1]=f1; o10[2]=f2;
    #pragma unroll
    for (int pI = 0; pI < 3; ++pI) {
      int i0 = PRS[pI][0], i1 = PRS[pI][1];
      float de = eta[i0] - eta[i1];
      float dp = wrap_dphi(phi[i0] - phi[i1]);
      o10[3 + pI] = sqrtf(de*de + dp*dp + 1e-8f);
      int j0 = J[i0], j1 = J[i1];
      float e = E[j0]+E[j1], X = px[j0]+px[j1], Y = py[j0]+py[j1], Z = pz[j0]+pz[j1];
      float m2ij = fmaxf(e*e - (X*X + Y*Y + Z*Z), 1e-8f);
      o10[7 + pI] = m2ij / m2g;
    }
    o10[6] = log1pf(m);
    _Float16* ip = intra + ((b * NG_) + g * NA_ + a) * (long)KPAD_PHYS;
    #pragma unroll
    for (int i = 0; i < 10; ++i) { ip[i] = (_Float16)o10[i]; feats[6 + g*10 + i] = o10[i]; }
    #pragma unroll
    for (int i = 10; i < KPAD_PHYS; ++i) ip[i] = (_Float16)0.f;
  }
  float m2a = fmaxf(P[0][0]*P[0][0] - (P[0][1]*P[0][1]+P[0][2]*P[0][2]+P[0][3]*P[0][3]), 1e-8f);
  float m2b = fmaxf(P[1][0]*P[1][0] - (P[1][1]*P[1][1]+P[1][2]*P[1][2]+P[1][3]*P[1][3]), 1e-8f);
  float m1 = sqrtf(m2a), m2 = sqrtf(m2b);
  float pt1 = fmaxf(sqrtf(P[0][1]*P[0][1] + P[0][2]*P[0][2]), 1e-8f);
  float pt2 = fmaxf(sqrtf(P[1][1]*P[1][1] + P[1][2]*P[1][2]), 1e-8f);
  float eta1 = asinhf(P[0][3] / pt1), eta2 = asinhf(P[1][3] / pt2);
  float phi1 = atan2f(P[0][2], P[0][1]), phi2 = atan2f(P[1][2], P[1][1]);
  float dphi = wrap_dphi(phi1 - phi2);
  float dr = sqrtf((eta1-eta2)*(eta1-eta2) + dphi*dphi + 1e-8f);
  feats[0] = log1pf(m1);
  feats[1] = log1pf(m2);
  feats[2] = fabsf(m1 - m2) / fmaxf(m1 + m2, 1e-8f);
  feats[3] = dr;
  feats[4] = (pt1 - pt2) / fmaxf(pt1 + pt2, 1e-8f);
  feats[5] = fabsf(dphi);
  float mu = 0.f;
  #pragma unroll
  for (int i = 0; i < 26; ++i) mu += feats[i];
  mu *= (1.f / 26.f);
  float var = 0.f;
  #pragma unroll
  for (int i = 0; i < 26; ++i) { float d = feats[i] - mu; var += d * d; }
  var *= (1.f / 26.f);
  float rstd = rsqrtf(var + 1e-5f);
  float* pp = phys26 + t * 26;
  #pragma unroll
  for (int i = 0; i < 26; ++i) pp[i] = png[i] * (feats[i] - mu) * rstd + pnb[i];
}

// ---------------------------------------------------------------------------
// ctx = mean over jets of final encoder x
// ---------------------------------------------------------------------------
__global__ __launch_bounds__(256)
void k_ctx(const float* __restrict__ x, float* __restrict__ ctx)
{
  long i = (long)blockIdx.x * blockDim.x + threadIdx.x;
  if (i >= (long)BATCH_ * DM) return;
  int d = (int)(i % DM); long b = i / DM;
  float s = 0.f;
  #pragma unroll
  for (int j = 0; j < SJ; ++j) s += x[(b * SJ + j) * (long)DM + d];
  ctx[i] = s * (1.f / 6.f);
}

// ---------------------------------------------------------------------------
// Build group sequences (chunk): tokens 0..2 = jet embeddings, token 3 = tok
// ---------------------------------------------------------------------------
__global__ __launch_bounds__(256)
void k_build_seqs(const float* __restrict__ x, const _Float16* __restrict__ tok,
                  float* __restrict__ seqs, long b0, long nb)
{
  long i = (long)blockIdx.x * blockDim.x + threadIdx.x;
  long total = nb * NG_ * GS_ * DM;
  if (i >= total) return;
  int d = (int)(i % DM); long r = i / DM;
  int p = (int)(r % GS_); long r2 = r / GS_;
  int s = (int)(r2 % NG_); long bl = r2 / NG_;
  long b = b0 + bl;
  float v;
  if (p < 3) v = x[(b * SJ + JET_TABLE[s][p]) * (long)DM + d];
  else       v = (float)tok[(b * NG_ + s) * (long)DM + d];
  seqs[r * (long)DM + d] = v;
}

// ---------------------------------------------------------------------------
// Mean-pool over the 4 group tokens (chunk) into pooled
// ---------------------------------------------------------------------------
__global__ __launch_bounds__(256)
void k_pool4(const float* __restrict__ seqs, float* __restrict__ pooled,
             long b0, long nb)
{
  long i = (long)blockIdx.x * blockDim.x + threadIdx.x;
  long total = nb * NG_ * DM;
  if (i >= total) return;
  int d = (int)(i % DM); long r = i / DM;
  int s = (int)(r % NG_); long bl = r / NG_;
  float sum = 0.f;
  #pragma unroll
  for (int p = 0; p < GS_; ++p)
    sum += seqs[((bl * NG_ + s) * (long)GS_ + p) * DM + d];
  pooled[((b0 + bl) * NG_ + s) * (long)DM + d] = sum * 0.25f;
}

// ---------------------------------------------------------------------------
// Build padded f16 feature rows [r1 | r2 | ctx | phys26 | 0-pad]
// ---------------------------------------------------------------------------
__global__ __launch_bounds__(256)
void k_feat(const float* __restrict__ pooled, const float* __restrict__ ctx,
            const float* __restrict__ phys26, _Float16* __restrict__ feat)
{
  long i = (long)blockIdx.x * blockDim.x + threadIdx.x;
  long total = (long)BATCH_ * NA_ * FEATP;
  if (i >= total) return;
  int c = (int)(i % FEATP); long r = i / FEATP;
  int a = (int)(r % NA_); long b = r / NA_;
  float v;
  if      (c < DM)     v = pooled[(b * NG_ + a) * (long)DM + c];
  else if (c < 2*DM)   v = pooled[(b * NG_ + NA_ + a) * (long)DM + (c - DM)];
  else if (c < 3*DM)   v = ctx[b * (long)DM + (c - 2*DM)];
  else if (c < FEATD)  v = phys26[r * 26 + (c - 3*DM)];
  else                 v = 0.f;
  feat[i] = (_Float16)v;
}

// ---------------------------------------------------------------------------
// Final N=1 projection: out[r] = h2[r,:] . w3 + b3
// ---------------------------------------------------------------------------
__global__ __launch_bounds__(256)
void k_score3(const _Float16* __restrict__ h2, const float* __restrict__ w3,
              const float* __restrict__ b3, float* __restrict__ out)
{
  long r = (long)blockIdx.x * blockDim.x + threadIdx.x;
  if (r >= (long)BATCH_ * NA_) return;
  float s = 0.f;
  #pragma unroll 8
  for (int d = 0; d < DM; ++d) s += (float)h2[r * (long)DM + d] * w3[d];
  out[r] = s + b3[0];
}

// ---------------------------------------------------------------------------
// Host-side orchestration
// ---------------------------------------------------------------------------
static inline long cdivl(long a, long b) { return (a + b - 1) / b; }

static void gemm_h(hipStream_t st, const _Float16* A, int lda,
                   const _Float16* W, int ldw, const float* bias,
                   _Float16* C, int ldc, long M, int N, int K, int act)
{
  long tasks = (M >> 4) * (long)(N >> 7);
  dim3 g((unsigned)cdivl(tasks, 8)), blk(256);
  switch (act) {
    case 0:  k_gemm<0,false,_Float16><<<g,blk,0,st>>>(A,lda,W,ldw,bias,nullptr,C,ldc,M,N,K); break;
    case 1:  k_gemm<1,false,_Float16><<<g,blk,0,st>>>(A,lda,W,ldw,bias,nullptr,C,ldc,M,N,K); break;
    default: k_gemm<2,false,_Float16><<<g,blk,0,st>>>(A,lda,W,ldw,bias,nullptr,C,ldc,M,N,K); break;
  }
}
static void gemm_resid(hipStream_t st, const _Float16* A, int lda,
                       const _Float16* W, int ldw, const float* bias,
                       const float* R, float* C, int ldc, long M, int N, int K)
{
  long tasks = (M >> 4) * (long)(N >> 7);
  dim3 g((unsigned)cdivl(tasks, 8)), blk(256);
  k_gemm<0,true,float><<<g,blk,0,st>>>(A,lda,W,ldw,bias,R,C,ldc,M,N,K);
}
static void cvt(hipStream_t st, const float* src, _Float16* dst,
                int rows, int ks, int kd)
{
  long total = (long)rows * kd;
  k_cvt_pad<<<dim3((unsigned)cdivl(total, 256)), dim3(256), 0, st>>>(src, dst, rows, ks, kd);
}

struct EncW {
  const float *ln1g, *ln1b, *qkvb, *outb, *ln2g, *ln2b, *ff1b, *ff2b;
  const float *qkvWf, *outWf, *ff1Wf, *ff2Wf;   // f32 source weights
  _Float16 *qkvW, *outW, *ff1W, *ff2W;           // f16 converted
};

extern "C" void kernel_launch(void* const* d_in, const int* in_sizes, int n_in,
                              void* d_out, int out_size, void* d_ws, size_t ws_size,
                              hipStream_t stream)
{
  (void)in_sizes; (void)n_in; (void)out_size; (void)ws_size;
  const float* fm   = (const float*)d_in[0];
  const float* WinW = (const float*)d_in[1];
  const float* WinB = (const float*)d_in[2];
  const float* pos  = (const float*)d_in[3];
  const float* ptw  = (const float*)d_in[4];

  EncW enc[5];  // [0..3] main layers, [4] group layer
  for (int L = 0; L < 5; ++L) {
    int o = 5 + 12 * L;
    enc[L].ln1g  = (const float*)d_in[o+0];  enc[L].ln1b = (const float*)d_in[o+1];
    enc[L].qkvWf = (const float*)d_in[o+2];  enc[L].qkvb = (const float*)d_in[o+3];
    enc[L].outWf = (const float*)d_in[o+4];  enc[L].outb = (const float*)d_in[o+5];
    enc[L].ln2g  = (const float*)d_in[o+6];  enc[L].ln2b = (const float*)d_in[o+7];
    enc[L].ff1Wf = (const float*)d_in[o+8];  enc[L].ff1b = (const float*)d_in[o+9];
    enc[L].ff2Wf = (const float*)d_in[o+10]; enc[L].ff2b = (const float*)d_in[o+11];
  }
  const float* physWf = (const float*)d_in[65];
  const float* physB  = (const float*)d_in[66];
  const float* pnG    = (const float*)d_in[67];
  const float* pnB    = (const float*)d_in[68];
  const float* s1Wf   = (const float*)d_in[69];
  const float* s1B    = (const float*)d_in[70];
  const float* s2Wf   = (const float*)d_in[71];
  const float* s2B    = (const float*)d_in[72];
  const float* s3W    = (const float*)d_in[73];
  const float* s3B    = (const float*)d_in[74];
  float* out = (float*)d_out;

  // ---- workspace layout (bump allocation, 256B aligned) ----
  char* p = (char*)d_ws;
  auto allocF = [&](size_t n) {
    float* r = (float*)p; p += ((n * sizeof(float) + 255) & ~(size_t)255); return r;
  };
  auto allocH = [&](size_t n) {
    _Float16* r = (_Float16*)p; p += ((n * sizeof(_Float16) + 255) & ~(size_t)255); return r;
  };
  float*    x      = allocF((size_t)BATCH_ * SJ * DM);
  float*    logpt  = allocF((size_t)BATCH_ * SJ);
  float*    ctxb   = allocF((size_t)BATCH_ * DM);
  float*    phys26 = allocF((size_t)BATCH_ * NA_ * 26);
  float*    pooled = allocF((size_t)BATCH_ * NG_ * DM);
  _Float16* intra  = allocH((size_t)BATCH_ * NG_ * KPAD_PHYS);
  _Float16* tok    = allocH((size_t)BATCH_ * NG_ * DM);
  // converted f16 weights
  for (int L = 0; L < 5; ++L) {
    enc[L].qkvW = allocH((size_t)3*DM * DM);
    enc[L].outW = allocH((size_t)DM * DM);
    enc[L].ff1W = allocH((size_t)DFF_ * DM);
    enc[L].ff2W = allocH((size_t)DM * DFF_);
  }
  _Float16* physW16 = allocH((size_t)DM * KPAD_PHYS);
  _Float16* s1W16   = allocH((size_t)2*DM * FEATP);
  _Float16* s2W16   = allocH((size_t)DM * 2*DM);
  char* arena0 = p;

  const long TOK = (long)BATCH_ * SJ;   // 49152 token rows
  const dim3 T256(256);

  // ==== one-time weight conversion / padding ====
  for (int L = 0; L < 5; ++L) {
    cvt(stream, enc[L].qkvWf, enc[L].qkvW, 3*DM, DM, DM);
    cvt(stream, enc[L].outWf, enc[L].outW, DM, DM, DM);
    cvt(stream, enc[L].ff1Wf, enc[L].ff1W, DFF_, DM, DM);
    cvt(stream, enc[L].ff2Wf, enc[L].ff2W, DM, DFF_, DFF_);
  }
  cvt(stream, physWf, physW16, DM, NA_, KPAD_PHYS);
  cvt(stream, s1Wf, s1W16, 2*DM, FEATD, FEATP);
  cvt(stream, s2Wf, s2W16, DM, 2*DM, 2*DM);

  // ==== embedding ====
  k_embed<<<dim3((unsigned)cdivl(TOK * DM, 256)), T256, 0, stream>>>(
      fm, WinW, WinB, pos, x, logpt);

  // ==== main encoder (4 layers) ====
  {
    char* q = arena0;
    auto sal = [&](size_t nh) {
      _Float16* r = (_Float16*)q; q += ((nh * 2 + 255) & ~(size_t)255); return r;
    };
    _Float16* y    = sal((size_t)TOK * DM);
    _Float16* qkvB = sal((size_t)TOK * 3 * DM);
    _Float16* ffh  = sal((size_t)TOK * DFF_);
    _Float16* att  = y;  // reuse: y consumed by qkv GEMM before attention
    for (int L = 0; L < 4; ++L) {
      k_ln128<<<dim3((unsigned)cdivl(TOK, 8)), T256, 0, stream>>>(
          x, enc[L].ln1g, enc[L].ln1b, y, TOK);
      gemm_h(stream, y, DM, enc[L].qkvW, DM, enc[L].qkvb, qkvB, 3*DM,
             TOK, 3*DM, DM, 0);
      k_attn<SJ, NHEAD_, HD_, true><<<dim3((unsigned)cdivl((long)BATCH_*NHEAD_*SJ, 256)),
          T256, 0, stream>>>(qkvB, logpt, ptw, att, BATCH_);
      gemm_resid(stream, att, DM, enc[L].outW, DM, enc[L].outb, x, x, DM,
                 TOK, DM, DM);
      k_ln128<<<dim3((unsigned)cdivl(TOK, 8)), T256, 0, stream>>>(
          x, enc[L].ln2g, enc[L].ln2b, y, TOK);
      gemm_h(stream, y, DM, enc[L].ff1W, DM, enc[L].ff1b, ffh, DFF_,
             TOK, DFF_, DM, 1);
      gemm_resid(stream, ffh, DFF_, enc[L].ff2W, DFF_, enc[L].ff2b, x, x, DM,
                 TOK, DM, DFF_);
    }
  }

  // ==== context + physics + phys tokens ====
  k_ctx<<<dim3((unsigned)cdivl((long)BATCH_ * DM, 256)), T256, 0, stream>>>(x, ctxb);
  k_physics<<<dim3((unsigned)cdivl((long)BATCH_ * NA_, 128)), dim3(128), 0, stream>>>(
      fm, pnG, pnB, intra, phys26);
  gemm_h(stream, intra, KPAD_PHYS, physW16, KPAD_PHYS, physB, tok, DM,
         (long)BATCH_ * NG_, DM, KPAD_PHYS, 2);   // gelu

  // ==== group encoder (layer index 4), chunked over batch ====
  {
    const EncW& g = enc[4];
    char* q = arena0;
    auto salF = [&](size_t n) {
      float* r = (float*)q; q += ((n * 4 + 255) & ~(size_t)255); return r;
    };
    auto salH = [&](size_t n) {
      _Float16* r = (_Float16*)q; q += ((n * 2 + 255) & ~(size_t)255); return r;
    };
    const long ROWS = (long)CB * NG_ * GS_;   // 81920 token rows per chunk
    float*    seqs = salF((size_t)ROWS * DM);
    _Float16* yg   = salH((size_t)ROWS * DM);
    _Float16* qg   = salH((size_t)ROWS * 3 * DM);
    _Float16* ffg  = qg;   // reuse qkv region for FF hidden (256 <= 384 wide)
    _Float16* attg = yg;
    for (int c = 0; c < NCHUNK; ++c) {
      long b0 = (long)c * CB;
      k_build_seqs<<<dim3((unsigned)cdivl(ROWS * DM, 256)), T256, 0, stream>>>(
          x, tok, seqs, b0, CB);
      k_ln128<<<dim3((unsigned)cdivl(ROWS, 8)), T256, 0, stream>>>(
          seqs, g.ln1g, g.ln1b, yg, ROWS);
      gemm_h(stream, yg, DM, g.qkvW, DM, g.qkvb, qg, 3*DM, ROWS, 3*DM, DM, 0);
      k_attn<GS_, GNH_, GHD_, false><<<dim3((unsigned)cdivl((long)CB*NG_*GNH_*GS_, 256)),
          T256, 0, stream>>>(qg, nullptr, nullptr, attg, (long)CB * NG_);
      gemm_resid(stream, attg, DM, g.outW, DM, g.outb, seqs, seqs, DM,
                 ROWS, DM, DM);
      k_ln128<<<dim3((unsigned)cdivl(ROWS, 8)), T256, 0, stream>>>(
          seqs, g.ln2g, g.ln2b, yg, ROWS);
      gemm_h(stream, yg, DM, g.ff1W, DM, g.ff1b, ffg, DFF_, ROWS, DFF_, DM, 1);
      gemm_resid(stream, ffg, DFF_, g.ff2W, DFF_, g.ff2b, seqs, seqs, DM,
                 ROWS, DM, DFF_);
      k_pool4<<<dim3((unsigned)cdivl((long)CB * NG_ * DM, 256)), T256, 0, stream>>>(
          seqs, pooled, b0, CB);
    }
  }

  // ==== scoring MLP ====
  {
    char* q = arena0;
    auto salH = [&](size_t n) {
      _Float16* r = (_Float16*)q; q += ((n * 2 + 255) & ~(size_t)255); return r;
    };
    const long SR = (long)BATCH_ * NA_;  // 81920 score rows
    _Float16* feat = salH((size_t)SR * FEATP);
    _Float16* h1   = salH((size_t)SR * 2*DM);
    _Float16* h2   = salH((size_t)SR * DM);
    k_feat<<<dim3((unsigned)cdivl(SR * FEATP, 256)), T256, 0, stream>>>(
        pooled, ctxb, phys26, feat);
    gemm_h(stream, feat, FEATP, s1W16, FEATP, s1B, h1, 2*DM, SR, 2*DM, FEATD + 6, 2);
    gemm_h(stream, h1, 2*DM, s2W16, 2*DM, s2B, h2, DM, SR, DM, 2*DM, 2);
    k_score3<<<dim3((unsigned)cdivl(SR, 256)), T256, 0, stream>>>(h2, s3W, s3B, out);
  }
}
